// SpatialSwinBlock_2499670966466
// MI455X (gfx1250) — matrix operand order repaired and checked by reading, not compiled, not run on hardware
//
#include <hip/hip_runtime.h>

// ---------------------------------------------------------------------------
// SwinV2 block for MI455X (gfx1250, wave32, WMMA f32_16x16x32_f16 + TDM)
// B=4 T=8 Hg=Wg=64 C=192 NH=6 WS=8 SS=4  -> N=64 tokens/window, HD=32,
// 2048 windows total, 131072 token rows.
// ---------------------------------------------------------------------------

typedef _Float16 f16;
typedef __attribute__((ext_vector_type(16))) _Float16 v16h;
typedef __attribute__((ext_vector_type(8)))  float    v8f;
typedef __attribute__((ext_vector_type(4)))  unsigned int v4u;
typedef __attribute__((ext_vector_type(8)))  int      v8i;
typedef __attribute__((ext_vector_type(4)))  int      v4i;

#define IMGS   32
#define CDIM   192
#define NHEADS 6
#define HDIM   32
#define NTOK   64
#define NWIN   2048
#define NROWS  131072
#define HID    768

#if __has_builtin(__builtin_amdgcn_tensor_load_to_lds) && \
    __has_builtin(__builtin_amdgcn_s_wait_tensorcnt)
#define USE_TDM 1
#else
#define USE_TDM 0
#endif

// --- WMMA fragment K-index helpers (per CDNA5 ISA 7.12.2 VGPR layouts) ------
// A (16x32 f16): lanes 0-15 row M=lane, K chunks {0..7,16..23};
//                lanes 16-31 same rows, K chunks {8..15,24..31}.
__device__ __forceinline__ int a_kbase(int vg, int lane) {
  int b = (vg < 4) ? (2 * vg) : (16 + 2 * (vg - 4));
  return b + ((lane & 16) ? 8 : 0);
}
// B (32x16 f16): lanes 0-15 col N=lane, K chunks {0..7,16..23}; lanes 16-31
// col N=lane-16, K chunks {8..15,24..31} (2 consecutive K per VGPR).
__device__ __forceinline__ int b_kbase(int vg, int lane) {
  int b = (vg < 4) ? (2 * vg) : (16 + 2 * (vg - 4));
  return b + ((lane & 16) ? 8 : 0);
}
__device__ __forceinline__ v8f v8f_zero() {
  v8f z;
#pragma unroll
  for (int i = 0; i < 8; ++i) z[i] = 0.0f;
  return z;
}
__device__ __forceinline__ v8f wmma16(v16h a, v16h b, v8f c) {
  return __builtin_amdgcn_wmma_f32_16x16x32_f16(false, a, false, b, (short)0,
                                                c, false, false);
}
// shifted-window mask region id along one axis (boundaries Hg-WS=56, Hg-SS=60)
__device__ __forceinline__ int rgn(int y) { return (y < 56) ? 0 : ((y < 60) ? 1 : 2); }

// low 32 bits of a generic pointer into __shared__ = LDS byte offset
__device__ __forceinline__ unsigned lds_off_of(const void* p) {
  return (unsigned)(unsigned long long)(uintptr_t)p;
}

// ---------------------------------------------------------------------------
// Tensor Data Mover: 2D f16 tile (tile_d0 x tile_d1), row stride = row length,
// LDS-padded +1 DWORD after every 64B (pad_interval=3, pad_amount=0) so rows
// land with a stride of 34 halves -> matches the [rows][34] LDS tiles.
// D# layout per CDNA5 ISA. 6-arg builtin form.
// ---------------------------------------------------------------------------
__device__ __forceinline__ void tdm_load_2d(unsigned lds_off, const void* gsrc,
                                            unsigned tile_d0, unsigned tile_d1,
                                            unsigned stride_elems) {
#if USE_TDM
  unsigned long long ga = (unsigned long long)(uintptr_t)gsrc;
  v4u g0;
  g0[0] = 1u;                                          // count=1 (valid, user)
  g0[1] = lds_off;                                     // lds_addr
  g0[2] = (unsigned)(ga & 0xffffffffu);                // global_addr[31:0]
  g0[3] = (unsigned)((ga >> 32) & 0x01ffffffu)         // global_addr[56:32]
          | (2u << 30);                                // type=2 ("image")
  unsigned td0 = stride_elems;                         // tensor_dim0 (row len)
  unsigned td1 = tile_d1;                              // tensor_dim1
  v8i g1;
  g1[0] = (int)((1u << 16)                             // data_size=1 -> 2 bytes
                | (1u << 20)                           // pad_enable
                | (3u << 22));                         // pad_interval=3 (64B)
  g1[1] = (int)((td0 & 0xffffu) << 16);                // tensor_dim0[15:0]
  g1[2] = (int)((td0 >> 16) | ((td1 & 0xffffu) << 16));// dim0[31:16]|dim1[15:0]
  g1[3] = (int)((td1 >> 16) | (tile_d0 << 16));        // dim1[31:16]|tile_dim0
  g1[4] = (int)tile_d1;                                // tile_dim1 (tile_dim2=0)
  g1[5] = (int)stride_elems;                           // tensor_dim0_stride lo
  g1[6] = 0;                                           // stride hi | dim1_stride lo
  g1[7] = 0;
  v4i g2;
  g2[0] = 0; g2[1] = 0; g2[2] = 0; g2[3] = 0;
  v4i g3;
  g3[0] = 0; g3[1] = 0; g3[2] = 0; g3[3] = 0;
  v8i g4;
#pragma unroll
  for (int i = 0; i < 8; ++i) g4[i] = 0;
  __builtin_amdgcn_tensor_load_to_lds(g0, g1, g2, g3, g4, 0);
#else
  (void)lds_off; (void)gsrc; (void)tile_d0; (void)tile_d1; (void)stride_elems;
#endif
}
__device__ __forceinline__ void tdm_wait0() {
#if USE_TDM
  __builtin_amdgcn_s_wait_tensorcnt((short)0);
#endif
}

// ---------------------------------------------------------------------------
// Kernel 0: weights -> f16  [qkv_w | proj_w | fc1_w | fc2_w]
// ---------------------------------------------------------------------------
#define WN_QKV  (576 * 192)
#define WN_PROJ (192 * 192)
#define WN_FC1  (768 * 192)
#define WN_FC2  (192 * 768)
#define WN_ALL  (WN_QKV + WN_PROJ + WN_FC1 + WN_FC2)

__global__ __launch_bounds__(256) void w2h_kernel(const float* __restrict__ qkv_w,
                                                  const float* __restrict__ proj_w,
                                                  const float* __restrict__ fc1_w,
                                                  const float* __restrict__ fc2_w,
                                                  f16* __restrict__ wh) {
  int i = blockIdx.x * 256 + threadIdx.x;
  if (i >= WN_ALL) return;
  float v;
  if (i < WN_QKV) v = qkv_w[i];
  else if (i < WN_QKV + WN_PROJ) v = proj_w[i - WN_QKV];
  else if (i < WN_QKV + WN_PROJ + WN_FC1) v = fc1_w[i - WN_QKV - WN_PROJ];
  else v = fc2_w[i - WN_QKV - WN_PROJ - WN_FC1];
  wh[i] = (f16)v;
}

// ---------------------------------------------------------------------------
// Kernel 1: continuous-position-bias MLP -> rpb[h][225] = 16*sigmoid(mlp(tbl))
// ---------------------------------------------------------------------------
__global__ __launch_bounds__(256) void cpb_kernel(const float* __restrict__ w1,
                                                  const float* __restrict__ b1,
                                                  const float* __restrict__ w2,
                                                  float* __restrict__ rpb) {
  int p = threadIdx.x;
  if (p >= 225) return;
  int a = p / 15, b = p % 15;
  float t0 = (float)(a - 7) / 7.0f * 8.0f;
  float t1 = (float)(b - 7) / 7.0f * 8.0f;
  const float inv_log2_8 = 1.0f / 3.0f;  // 1/log2(8)
  t0 = copysignf(log2f(fabsf(t0) + 1.0f), t0) * inv_log2_8;
  t1 = copysignf(log2f(fabsf(t1) + 1.0f), t1) * inv_log2_8;
  float acc[NHEADS];
#pragma unroll
  for (int h = 0; h < NHEADS; ++h) acc[h] = 0.0f;
  for (int j = 0; j < 512; ++j) {
    float hv = fmaxf(w1[2 * j] * t0 + w1[2 * j + 1] * t1 + b1[j], 0.0f);
#pragma unroll
    for (int h = 0; h < NHEADS; ++h) acc[h] += w2[h * 512 + j] * hv;
  }
#pragma unroll
  for (int h = 0; h < NHEADS; ++h)
    rpb[h * 225 + p] = 16.0f / (1.0f + __expf(-acc[h]));
}

// ---------------------------------------------------------------------------
// Kernel 2: QKV GEMM per window: [64 x 192] @ [192 x 576] + bias
// roll(-4,-4) + window partition folded into the A-tile load addressing.
// B tile staged by the Tensor Data Mover from pre-converted f16 weights.
// qkv_out layout: [win][s(3)][h(6)][tok(64)][d(32)] f32
// ---------------------------------------------------------------------------
__global__ __launch_bounds__(512) void qkv_kernel(const float* __restrict__ x,
                                                  const f16* __restrict__ wh_qkv,
                                                  const float* __restrict__ q_bias,
                                                  const float* __restrict__ v_bias,
                                                  float* __restrict__ qkv_out) {
  __shared__ f16 Xs[64][34];
  __shared__ f16 Ws[576][34];
  const int w = blockIdx.x;
  const int img = w >> 6, wy = (w >> 3) & 7, wx = w & 7;
  const int tid = threadIdx.x;
  const int wave = tid >> 5, lane = tid & 31;
  const int mt = wave & 3;          // wave's M tile (stride-16 tile ids => fixed)
  const int col0 = lane & 15;
  const int hi = (lane >> 4) & 1;

  v8f acc[9];
#pragma unroll
  for (int i = 0; i < 9; ++i) acc[i] = v8f_zero();

  for (int kk = 0; kk < 6; ++kk) {
    const int K0 = kk * 32;
#if USE_TDM
    if (wave == 0) {
      tdm_load_2d(lds_off_of(&Ws[0][0]), wh_qkv + K0, 32, 576, CDIM);
    }
#else
    for (int e = tid; e < 576 * 32; e += 512) {
      int n = e >> 5, c = e & 31;
      Ws[n][c] = wh_qkv[n * CDIM + K0 + c];
    }
#endif
    // stage A: window tokens (rolled frame -> original pixel), f32 -> f16
    for (int e = tid; e < 64 * 32; e += 512) {
      int r = e >> 5, c = e & 31;
      int ty = r >> 3, tx = r & 7;
      int y = (wy * 8 + ty + 4) & 63;
      int xc = (wx * 8 + tx + 4) & 63;
      size_t row = (size_t)img * 4096 + (size_t)(y * 64 + xc);
      Xs[r][c] = (f16)x[row * CDIM + K0 + c];
    }
#if USE_TDM
    if (wave == 0) tdm_wait0();
#endif
    __syncthreads();

    v16h a;
    const int arow = mt * 16 + col0;
#pragma unroll
    for (int vg = 0; vg < 8; ++vg) {
      int kb = a_kbase(vg, lane);
      a[2 * vg]     = Xs[arow][kb];
      a[2 * vg + 1] = Xs[arow][kb + 1];
    }
#pragma unroll
    for (int i = 0; i < 9; ++i) {
      const int nt = (wave >> 2) + 4 * i;
      const int bn = nt * 16 + col0;
      v16h bf;
#pragma unroll
      for (int vg = 0; vg < 8; ++vg) {
        int kb = b_kbase(vg, lane);
        bf[2 * vg]     = Ws[bn][kb];
        bf[2 * vg + 1] = Ws[bn][kb + 1];
      }
      acc[i] = wmma16(a, bf, acc[i]);
    }
    __syncthreads();
  }

  // epilogue: + concat(q_bias, 0, v_bias), scatter to [w][s][h][tok][d]
#pragma unroll
  for (int i = 0; i < 9; ++i) {
    const int nt = (wave >> 2) + 4 * i;
    const int n = nt * 16 + col0;
    const int s = n / CDIM, c = n % CDIM;
    const int h = c >> 5, d = c & 31;
    float bias = (s == 0) ? q_bias[c] : ((s == 2) ? v_bias[c] : 0.0f);
    size_t base = ((((size_t)w * 3 + s) * NHEADS + h) * 2048) + d;
#pragma unroll
    for (int vg = 0; vg < 8; ++vg) {
      int r = mt * 16 + vg + hi * 8;
      qkv_out[base + (size_t)r * 32] = acc[i][vg] + bias;
    }
  }
}

// ---------------------------------------------------------------------------
// Kernel 3: cosine attention per (window, head). One wave per unit.
//  S = (qn*scale) @ kn^T (+cpb bias, +shift mask), softmax, O = P @ V.
// ---------------------------------------------------------------------------
__global__ __launch_bounds__(64) void attn_kernel(const float* __restrict__ qkv,
                                                  const float* __restrict__ rpb,
                                                  const float* __restrict__ flex,
                                                  f16* __restrict__ attn_out) {
  // per-wave: q 64x32, k 64x32, v 64x32, P 64x64 (all f16) = 10240 halves
  __shared__ f16 smem[2][10240];
  const int wave = threadIdx.x >> 5, lane = threadIdx.x & 31;
  const int unit = blockIdx.x * 2 + wave;
  const int w = unit / NHEADS, h = unit % NHEADS;
  const int wy = (w >> 3) & 7, wx = w & 7;
  const int col0 = lane & 15, hi = (lane >> 4) & 1;

  f16* qs = smem[wave];
  f16* ks = qs + 2048;
  f16* vs = ks + 2048;
  f16* Ps = vs + 2048;

  const float scale_h = __expf(fminf(flex[h], 4.6051702f));  // min(.., log 100)
  const float* qp = qkv + (((size_t)w * 3 + 0) * NHEADS + h) * 2048;
  const float* kp = qkv + (((size_t)w * 3 + 1) * NHEADS + h) * 2048;
  const float* vp = qkv + (((size_t)w * 3 + 2) * NHEADS + h) * 2048;

  // ---- load + cosine-normalize (scale folded into q) ----
  for (int r = lane; r < 64; r += 32) {
    const float* p = qp + r * 32;
    float ss = 0.0f;
#pragma unroll
    for (int d = 0; d < 32; ++d) { float v = p[d]; ss += v * v; }
    float sc = scale_h / fmaxf(sqrtf(ss), 1e-12f);
#pragma unroll
    for (int d = 0; d < 32; ++d) qs[r * 32 + d] = (f16)(p[d] * sc);

    p = kp + r * 32;
    ss = 0.0f;
#pragma unroll
    for (int d = 0; d < 32; ++d) { float v = p[d]; ss += v * v; }
    sc = 1.0f / fmaxf(sqrtf(ss), 1e-12f);
#pragma unroll
    for (int d = 0; d < 32; ++d) ks[r * 32 + d] = (f16)(p[d] * sc);

    p = vp + r * 32;
#pragma unroll
    for (int d = 0; d < 32; ++d) vs[r * 32 + d] = (f16)p[d];
  }
  __syncthreads();

  // ---- S = q @ k^T (+bias +mask), row softmax per 16-row block ----
  for (int mt = 0; mt < 4; ++mt) {
    v16h a;
    const int arow = mt * 16 + col0;
#pragma unroll
    for (int vg = 0; vg < 8; ++vg) {
      int kb = a_kbase(vg, lane);
      a[2 * vg]     = qs[arow * 32 + kb];
      a[2 * vg + 1] = qs[arow * 32 + kb + 1];
    }
    float sreg[4][8];
#pragma unroll
    for (int nt = 0; nt < 4; ++nt) {
      const int j = nt * 16 + col0;  // key token (column of S)
      v16h bf;
#pragma unroll
      for (int vg = 0; vg < 8; ++vg) {
        int kb = b_kbase(vg, lane);
        bf[2 * vg]     = ks[j * 32 + kb];
        bf[2 * vg + 1] = ks[j * 32 + kb + 1];
      }
      v8f c = v8f_zero();
      c = wmma16(a, bf, c);
      const int yj = j >> 3, xj = j & 7;
      const int rj = rgn(wy * 8 + yj) * 3 + rgn(wx * 8 + xj);
#pragma unroll
      for (int vg = 0; vg < 8; ++vg) {
        int i = mt * 16 + vg + hi * 8;  // query token (row of S)
        int yi = i >> 3, xi = i & 7;
        int ri = rgn(wy * 8 + yi) * 3 + rgn(wx * 8 + xi);
        int dyx = (yi - yj + 7) * 15 + (xi - xj + 7);
        float s = c[vg] + rpb[h * 225 + dyx];
        if (ri != rj) s -= 100.0f;
        sreg[nt][vg] = s;
      }
    }
    // softmax: each row lives in one VGPR slot across a 16-lane half
#pragma unroll
    for (int vg = 0; vg < 8; ++vg) {
      float m = sreg[0][vg];
#pragma unroll
      for (int nt = 1; nt < 4; ++nt) m = fmaxf(m, sreg[nt][vg]);
#pragma unroll
      for (int off = 1; off < 16; off <<= 1) m = fmaxf(m, __shfl_xor(m, off, 16));
      float sum = 0.0f;
#pragma unroll
      for (int nt = 0; nt < 4; ++nt) {
        float p = __expf(sreg[nt][vg] - m);
        sreg[nt][vg] = p;
        sum += p;
      }
#pragma unroll
      for (int off = 1; off < 16; off <<= 1) sum += __shfl_xor(sum, off, 16);
      float inv = 1.0f / sum;
      int i = mt * 16 + vg + hi * 8;
#pragma unroll
      for (int nt = 0; nt < 4; ++nt)
        Ps[i * 64 + nt * 16 + col0] = (f16)(sreg[nt][vg] * inv);
    }
  }
  __syncthreads();

  // ---- O = P @ V ----
  for (int mt = 0; mt < 4; ++mt) {
#pragma unroll
    for (int dt = 0; dt < 2; ++dt) {
      v8f c = v8f_zero();
#pragma unroll
      for (int kc = 0; kc < 2; ++kc) {
        v16h a, bf;
        const int arow = mt * 16 + col0;
#pragma unroll
        for (int vg = 0; vg < 8; ++vg) {
          int kb = kc * 32 + a_kbase(vg, lane);
          a[2 * vg]     = Ps[arow * 64 + kb];
          a[2 * vg + 1] = Ps[arow * 64 + kb + 1];
        }
        const int nd = dt * 16 + col0;
#pragma unroll
        for (int vg = 0; vg < 8; ++vg) {
          int kb = kc * 32 + b_kbase(vg, lane);
          bf[2 * vg]     = vs[kb * 32 + nd];
          bf[2 * vg + 1] = vs[(kb + 1) * 32 + nd];
        }
        c = wmma16(a, bf, c);
      }
#pragma unroll
      for (int vg = 0; vg < 8; ++vg) {
        int i = mt * 16 + vg + hi * 8;
        int cc = h * HDIM + dt * 16 + col0;
        attn_out[((size_t)w * 64 + i) * CDIM + cc] = (f16)c[vg];
      }
    }
  }
}

// ---------------------------------------------------------------------------
// Kernel 4: proj GEMM [64x192]@[192x192] + bias, window-reverse + roll(+4),
// x1 = shortcut + LN1(out). 4 waves; wave owns 16 full rows -> shuffle LN.
// A and B tiles staged by the Tensor Data Mover.
// ---------------------------------------------------------------------------
__global__ __launch_bounds__(128) void proj_kernel(const f16* __restrict__ attn_out,
                                                   const f16* __restrict__ wh_proj,
                                                   const float* __restrict__ proj_b,
                                                   const float* __restrict__ x_in,
                                                   const float* __restrict__ g1,
                                                   const float* __restrict__ bb1,
                                                   float* __restrict__ x1,
                                                   f16* __restrict__ x1h) {
  __shared__ f16 As[64][34];
  __shared__ f16 Bs[192][34];
  const int w = blockIdx.x;
  const int img = w >> 6, wy = (w >> 3) & 7, wx = w & 7;
  const int tid = threadIdx.x;
  const int wave = tid >> 5, lane = tid & 31;
  const int col0 = lane & 15, hi = (lane >> 4) & 1;

  v8f acc[12];
#pragma unroll
  for (int i = 0; i < 12; ++i) acc[i] = v8f_zero();

  for (int kk = 0; kk < 6; ++kk) {
    const int K0 = kk * 32;
#if USE_TDM
    if (wave == 0) {
      tdm_load_2d(lds_off_of(&As[0][0]), attn_out + (size_t)w * 64 * CDIM + K0,
                  32, 64, CDIM);
      tdm_load_2d(lds_off_of(&Bs[0][0]), wh_proj + K0, 32, 192, CDIM);
      tdm_wait0();
    }
#else
    for (int e = tid; e < 64 * 32; e += 128) {
      int r = e >> 5, c = e & 31;
      As[r][c] = attn_out[((size_t)w * 64 + r) * CDIM + K0 + c];
    }
    for (int e = tid; e < 192 * 32; e += 128) {
      int n = e >> 5, c = e & 31;
      Bs[n][c] = wh_proj[n * CDIM + K0 + c];
    }
#endif
    __syncthreads();
    v16h a;
    const int arow = wave * 16 + col0;
#pragma unroll
    for (int vg = 0; vg < 8; ++vg) {
      int kb = a_kbase(vg, lane);
      a[2 * vg] = As[arow][kb];
      a[2 * vg + 1] = As[arow][kb + 1];
    }
#pragma unroll
    for (int nt = 0; nt < 12; ++nt) {
      const int bn = nt * 16 + col0;
      v16h bf;
#pragma unroll
      for (int vg = 0; vg < 8; ++vg) {
        int kb = b_kbase(vg, lane);
        bf[2 * vg] = Bs[bn][kb];
        bf[2 * vg + 1] = Bs[bn][kb + 1];
      }
      acc[nt] = wmma16(a, bf, acc[nt]);
    }
    __syncthreads();
  }

  // epilogue: bias, LayerNorm over the 192 cols of each row, residual
#pragma unroll
  for (int vg = 0; vg < 8; ++vg) {
    float o[12], sum = 0.0f, sq = 0.0f;
#pragma unroll
    for (int nt = 0; nt < 12; ++nt) {
      int n = nt * 16 + col0;
      float v = acc[nt][vg] + proj_b[n];
      o[nt] = v;
      sum += v;
      sq += v * v;
    }
#pragma unroll
    for (int off = 1; off < 16; off <<= 1) {
      sum += __shfl_xor(sum, off, 16);
      sq  += __shfl_xor(sq, off, 16);
    }
    float mean = sum * (1.0f / 192.0f);
    float var = sq * (1.0f / 192.0f) - mean * mean;
    float rstd = rsqrtf(var + 1e-5f);
    int i = wave * 16 + vg + hi * 8;
    int ty = i >> 3, tx = i & 7;
    int y = (wy * 8 + ty + 4) & 63;
    int xc = (wx * 8 + tx + 4) & 63;
    size_t row = (size_t)img * 4096 + (size_t)(y * 64 + xc);
#pragma unroll
    for (int nt = 0; nt < 12; ++nt) {
      int n = nt * 16 + col0;
      float v = x_in[row * CDIM + n] + (o[nt] - mean) * rstd * g1[n] + bb1[n];
      x1[row * CDIM + n] = v;
      x1h[row * CDIM + n] = (f16)v;
    }
  }
}

// ---------------------------------------------------------------------------
// Kernel 5: fc1 GEMM [64x192]@[192x768] + bias + exact GELU -> f16 hidden
// grid = 2048 rowblocks * 4 colblocks; tiles staged by TDM.
// ---------------------------------------------------------------------------
__global__ __launch_bounds__(128) void fc1_kernel(const f16* __restrict__ x1h,
                                                  const f16* __restrict__ wh_fc1,
                                                  const float* __restrict__ fc1_b,
                                                  f16* __restrict__ h1) {
  __shared__ f16 As[64][34];
  __shared__ f16 Bs[192][34];
  const int rb = blockIdx.x >> 2, nb = blockIdx.x & 3;
  const int tid = threadIdx.x;
  const int wave = tid >> 5, lane = tid & 31;
  const int col0 = lane & 15, hi = (lane >> 4) & 1;

  v8f acc[12];
#pragma unroll
  for (int i = 0; i < 12; ++i) acc[i] = v8f_zero();

  for (int kk = 0; kk < 6; ++kk) {
    const int K0 = kk * 32;
#if USE_TDM
    if (wave == 0) {
      tdm_load_2d(lds_off_of(&As[0][0]), x1h + (size_t)rb * 64 * CDIM + K0,
                  32, 64, CDIM);
      tdm_load_2d(lds_off_of(&Bs[0][0]),
                  wh_fc1 + (size_t)nb * 192 * CDIM + K0, 32, 192, CDIM);
      tdm_wait0();
    }
#else
    for (int e = tid; e < 64 * 32; e += 128) {
      int r = e >> 5, c = e & 31;
      As[r][c] = x1h[((size_t)rb * 64 + r) * CDIM + K0 + c];
    }
    for (int e = tid; e < 192 * 32; e += 128) {
      int n = e >> 5, c = e & 31;
      Bs[n][c] = wh_fc1[(size_t)(nb * 192 + n) * CDIM + K0 + c];
    }
#endif
    __syncthreads();
    v16h a;
    const int arow = wave * 16 + col0;
#pragma unroll
    for (int vg = 0; vg < 8; ++vg) {
      int kb = a_kbase(vg, lane);
      a[2 * vg] = As[arow][kb];
      a[2 * vg + 1] = As[arow][kb + 1];
    }
#pragma unroll
    for (int nt = 0; nt < 12; ++nt) {
      const int bn = nt * 16 + col0;
      v16h bf;
#pragma unroll
      for (int vg = 0; vg < 8; ++vg) {
        int kb = b_kbase(vg, lane);
        bf[2 * vg] = Bs[bn][kb];
        bf[2 * vg + 1] = Bs[bn][kb + 1];
      }
      acc[nt] = wmma16(a, bf, acc[nt]);
    }
    __syncthreads();
  }

#pragma unroll
  for (int nt = 0; nt < 12; ++nt) {
    int n = nb * 192 + nt * 16 + col0;
    float bias = fc1_b[n];
#pragma unroll
    for (int vg = 0; vg < 8; ++vg) {
      int r = rb * 64 + wave * 16 + vg + hi * 8;
      float v = acc[nt][vg] + bias;
      float ge = 0.5f * v * (1.0f + erff(v * 0.70710678f));
      h1[(size_t)r * HID + n] = (f16)ge;
    }
  }
}

// ---------------------------------------------------------------------------
// Kernel 6: fc2 GEMM [64x768]@[768x192] + bias, out = x1 + LN2(h)
// tiles staged by TDM.
// ---------------------------------------------------------------------------
__global__ __launch_bounds__(128) void fc2_kernel(const f16* __restrict__ h1,
                                                  const f16* __restrict__ wh_fc2,
                                                  const float* __restrict__ fc2_b,
                                                  const float* __restrict__ x1,
                                                  const float* __restrict__ g2,
                                                  const float* __restrict__ bb2,
                                                  float* __restrict__ out) {
  __shared__ f16 As[64][34];
  __shared__ f16 Bs[192][34];
  const int rb = blockIdx.x;
  const int tid = threadIdx.x;
  const int wave = tid >> 5, lane = tid & 31;
  const int col0 = lane & 15, hi = (lane >> 4) & 1;

  v8f acc[12];
#pragma unroll
  for (int i = 0; i < 12; ++i) acc[i] = v8f_zero();

  for (int kk = 0; kk < 24; ++kk) {
    const int K0 = kk * 32;
#if USE_TDM
    if (wave == 0) {
      tdm_load_2d(lds_off_of(&As[0][0]), h1 + (size_t)rb * 64 * HID + K0,
                  32, 64, HID);
      tdm_load_2d(lds_off_of(&Bs[0][0]), wh_fc2 + K0, 32, 192, HID);
      tdm_wait0();
    }
#else
    for (int e = tid; e < 64 * 32; e += 128) {
      int r = e >> 5, c = e & 31;
      As[r][c] = h1[((size_t)rb * 64 + r) * HID + K0 + c];
    }
    for (int e = tid; e < 192 * 32; e += 128) {
      int n = e >> 5, c = e & 31;
      Bs[n][c] = wh_fc2[(size_t)n * HID + K0 + c];
    }
#endif
    __syncthreads();
    v16h a;
    const int arow = wave * 16 + col0;
#pragma unroll
    for (int vg = 0; vg < 8; ++vg) {
      int kb = a_kbase(vg, lane);
      a[2 * vg] = As[arow][kb];
      a[2 * vg + 1] = As[arow][kb + 1];
    }
#pragma unroll
    for (int nt = 0; nt < 12; ++nt) {
      const int bn = nt * 16 + col0;
      v16h bf;
#pragma unroll
      for (int vg = 0; vg < 8; ++vg) {
        int kb = b_kbase(vg, lane);
        bf[2 * vg] = Bs[bn][kb];
        bf[2 * vg + 1] = Bs[bn][kb + 1];
      }
      acc[nt] = wmma16(a, bf, acc[nt]);
    }
    __syncthreads();
  }

#pragma unroll
  for (int vg = 0; vg < 8; ++vg) {
    float o[12], sum = 0.0f, sq = 0.0f;
#pragma unroll
    for (int nt = 0; nt < 12; ++nt) {
      int n = nt * 16 + col0;
      float v = acc[nt][vg] + fc2_b[n];
      o[nt] = v;
      sum += v;
      sq += v * v;
    }
#pragma unroll
    for (int off = 1; off < 16; off <<= 1) {
      sum += __shfl_xor(sum, off, 16);
      sq  += __shfl_xor(sq, off, 16);
    }
    float mean = sum * (1.0f / 192.0f);
    float var = sq * (1.0f / 192.0f) - mean * mean;
    float rstd = rsqrtf(var + 1e-5f);
    size_t row = (size_t)rb * 64 + wave * 16 + vg + hi * 8;
#pragma unroll
    for (int nt = 0; nt < 12; ++nt) {
      int n = nt * 16 + col0;
      out[row * CDIM + n] =
          x1[row * CDIM + n] + (o[nt] - mean) * rstd * g2[n] + bb2[n];
    }
  }
}

// ---------------------------------------------------------------------------
// Host launcher
// ---------------------------------------------------------------------------
extern "C" void kernel_launch(void* const* d_in, const int* in_sizes, int n_in,
                              void* d_out, int out_size, void* d_ws, size_t ws_size,
                              hipStream_t stream) {
  const float* x       = (const float*)d_in[0];
  const float* qkv_w   = (const float*)d_in[1];
  const float* q_bias  = (const float*)d_in[2];
  const float* v_bias  = (const float*)d_in[3];
  const float* flex    = (const float*)d_in[4];
  const float* cpb_w1  = (const float*)d_in[5];
  const float* cpb_b1  = (const float*)d_in[6];
  const float* cpb_w2  = (const float*)d_in[7];
  const float* proj_w  = (const float*)d_in[8];
  const float* proj_b  = (const float*)d_in[9];
  const float* norm1_g = (const float*)d_in[10];
  const float* norm1_b = (const float*)d_in[11];
  const float* norm2_g = (const float*)d_in[12];
  const float* norm2_b = (const float*)d_in[13];
  const float* fc1_w   = (const float*)d_in[14];
  const float* fc1_b   = (const float*)d_in[15];
  const float* fc2_w   = (const float*)d_in[16];
  const float* fc2_b   = (const float*)d_in[17];
  float* out = (float*)d_out;

  char* ws = (char*)d_ws;
  const size_t OFF_RPB = 0;                                     // 6*225 f32
  const size_t OFF_WH  = 8192;                                  // f16 weights
  const size_t SZ_WH   = (size_t)WN_ALL * 2;                    // 884,736
  const size_t OFF_QKV = OFF_WH + SZ_WH;                        // [2048][3][6][64][32] f32
  const size_t SZ_QKV  = (size_t)NWIN * 3 * NHEADS * 2048 * 4;  // 301,989,888
  const size_t OFF_AO  = OFF_QKV + SZ_QKV;                      // [2048][64][192] f16
  const size_t SZ_AO   = (size_t)NWIN * 64 * CDIM * 2;          //  50,331,648
  const size_t OFF_X1  = OFF_AO + SZ_AO;                        // [131072][192] f32
  const size_t SZ_X1   = (size_t)NROWS * CDIM * 4;              // 100,663,296
  const size_t OFF_X1H = OFF_X1 + SZ_X1;                        // f16 copy
  const size_t SZ_X1H  = (size_t)NROWS * CDIM * 2;              //  50,331,648
  const size_t OFF_H1  = OFF_X1H + SZ_X1H;                      // [131072][768] f16

  float* ws_rpb = (float*)(ws + OFF_RPB);
  f16*   ws_wh  = (f16*)(ws + OFF_WH);
  float* ws_qkv = (float*)(ws + OFF_QKV);
  f16*   ws_ao  = (f16*)(ws + OFF_AO);
  float* ws_x1  = (float*)(ws + OFF_X1);
  f16*   ws_x1h = (f16*)(ws + OFF_X1H);
  f16*   ws_h1  = (f16*)(ws + OFF_H1);

  f16* wh_qkv  = ws_wh;
  f16* wh_proj = ws_wh + WN_QKV;
  f16* wh_fc1  = ws_wh + WN_QKV + WN_PROJ;
  f16* wh_fc2  = ws_wh + WN_QKV + WN_PROJ + WN_FC1;

  w2h_kernel<<<(WN_ALL + 255) / 256, 256, 0, stream>>>(qkv_w, proj_w, fc1_w,
                                                       fc2_w, ws_wh);
  cpb_kernel<<<1, 256, 0, stream>>>(cpb_w1, cpb_b1, cpb_w2, ws_rpb);
  qkv_kernel<<<NWIN, 512, 0, stream>>>(x, wh_qkv, q_bias, v_bias, ws_qkv);
  attn_kernel<<<(NWIN * NHEADS) / 2, 64, 0, stream>>>(ws_qkv, ws_rpb, flex, ws_ao);
  proj_kernel<<<NWIN, 128, 0, stream>>>(ws_ao, wh_proj, proj_b, x, norm1_g,
                                        norm1_b, ws_x1, ws_x1h);
  fc1_kernel<<<(NROWS / 64) * 4, 128, 0, stream>>>(ws_x1h, wh_fc1, fc1_b, ws_h1);
  fc2_kernel<<<NROWS / 64, 128, 0, stream>>>(ws_h1, wh_fc2, fc2_b, ws_x1,
                                             norm2_g, norm2_b, out);
}
